// gruaging_gru_70308614636165
// MI455X (gfx1250) — compile-verified, matrix operand order checked by reading, not run on hardware
//
#include <hip/hip_runtime.h>

typedef __bf16 bf16;
typedef __attribute__((ext_vector_type(16))) __bf16 v16bf;
typedef __attribute__((ext_vector_type(8)))  float  v8f;

namespace {
constexpr int kB  = 128;
constexpr int kT  = 1024;
constexpr int kI  = 64;
constexpr int kH  = 256;
constexpr int kHA = 512;
constexpr int kOUT = 16;
constexpr int kO1 = kH * kI;          // 16384
constexpr int kO2 = kO1 + kH * kH;    // 81920
constexpr int kO3 = kO2 + kH;         // 82176
constexpr int kP  = kO3 + kH;         // 82432
constexpr int kUP = 96;               // u padded: 65 -> 96 (K multiple of 32)
}

struct U4x2 { uint4 lo, hi; };
struct BF8  { bf16 v[8]; };

// ---- WMMA fragment helpers (CDNA5 16-bit A/B striping, wave32) -------------
// lanes 0-15 : row = lane,    K chunks {k0+0..7, k0+16..23}
// lanes 16-31: row = lane-16, K chunks {k0+8..15, k0+24..31}
static __device__ __forceinline__ v16bf ld_tile(const bf16* __restrict__ base,
                                                int row0, size_t ld, int k0, int lane) {
    const bf16* p = base + (size_t)(row0 + (lane & 15)) * ld + (size_t)(k0 + ((lane >> 4) << 3));
    U4x2 f;
    f.lo = *reinterpret_cast<const uint4*>(p);
    f.hi = *reinterpret_cast<const uint4*>(p + 16);
    return __builtin_bit_cast(v16bf, f);
}

static __device__ __forceinline__ v16bf ld_tile_f32(const float* __restrict__ base,
                                                    int row0, size_t ld, int k0, int lane) {
    const float* p = base + (size_t)(row0 + (lane & 15)) * ld + (size_t)(k0 + ((lane >> 4) << 3));
    bf16 t[16];
#pragma unroll
    for (int i = 0; i < 8; ++i) t[i] = (bf16)p[i];
#pragma unroll
    for (int i = 0; i < 8; ++i) t[8 + i] = (bf16)p[16 + i];
    return __builtin_bit_cast(v16bf, t);
}

static __device__ __forceinline__ v8f wmma_bf16(v16bf a, v16bf b, v8f c) {
    return __builtin_amdgcn_wmma_f32_16x16x32_bf16(false, a, false, b, (short)0, c, false, false);
}

static __device__ __forceinline__ float sigf(float x) {
    return 1.0f / (1.0f + __expf(-x));
}
static __device__ __forceinline__ float tanh_f(float x) {
    if (x > 15.f)  return 1.f;
    if (x < -15.f) return -1.f;
    float e = __expf(2.0f * x);
    return (e - 1.0f) / (e + 1.0f);
}

static __device__ __forceinline__ void dot8_bf_f32(const bf16* __restrict__ w,
                                                   const float* __restrict__ h, float& acc) {
    uint4 wq = *reinterpret_cast<const uint4*>(w);
    BF8 wb = __builtin_bit_cast(BF8, wq);
    float4 h0 = *reinterpret_cast<const float4*>(h);
    float4 h1 = *reinterpret_cast<const float4*>(h + 4);
    acc = fmaf((float)wb.v[0], h0.x, acc);
    acc = fmaf((float)wb.v[1], h0.y, acc);
    acc = fmaf((float)wb.v[2], h0.z, acc);
    acc = fmaf((float)wb.v[3], h0.w, acc);
    acc = fmaf((float)wb.v[4], h1.x, acc);
    acc = fmaf((float)wb.v[5], h1.y, acc);
    acc = fmaf((float)wb.v[6], h1.z, acc);
    acc = fmaf((float)wb.v[7], h1.w, acc);
}

static __device__ __forceinline__ void dot8_bf_bf(const bf16* __restrict__ w,
                                                  const bf16* __restrict__ x, float& acc) {
    uint4 wq = *reinterpret_cast<const uint4*>(w);
    uint4 xq = *reinterpret_cast<const uint4*>(x);
    BF8 wb = __builtin_bit_cast(BF8, wq);
    BF8 xb = __builtin_bit_cast(BF8, xq);
#pragma unroll
    for (int i = 0; i < 8; ++i) acc = fmaf((float)wb.v[i], (float)xb.v[i], acc);
}

// ---- conversion / padding ---------------------------------------------------
__global__ void k_conv_pad(const float* __restrict__ src, bf16* __restrict__ dst,
                           int rows, int scols, int dcols) {
    int idx = blockIdx.x * blockDim.x + threadIdx.x;
    if (idx >= rows * dcols) return;
    int r = idx / dcols, c = idx - r * dcols;
    dst[idx] = (c < scols) ? (bf16)src[(size_t)r * scols + c] : (bf16)0.0f;
}

// ---- Phase A: age-GRU one timestep -----------------------------------------
// grid 32 (HA col-tiles), block 256 (8 waves = 8 batch-row tiles)
__global__ __launch_bounds__(256) void k_age_step(
    const bf16* __restrict__ u_b, int t,
    const bf16* __restrict__ WihA_b, const bf16* __restrict__ WhhA_b,
    const float* __restrict__ bihA, const float* __restrict__ bhhA,
    const float* __restrict__ h_in, const bf16* __restrict__ h_in_b,
    float* __restrict__ h_out, bf16* __restrict__ h_out_b) {
    const int lane = threadIdx.x & 31;
    const int wave = threadIdx.x >> 5;
    const int m0 = wave * 16;                  // batch tile
    const int j0 = blockIdx.x * 16;            // hidden-col tile

    v8f cr = {}, cz = {}, cgin = {}, cghn = {};

    const bf16* xA = u_b + (size_t)t * kUP;    // row stride = T*kUP
#pragma unroll
    for (int k0 = 0; k0 < kUP; k0 += 32) {
        v16bf a  = ld_tile(xA, m0, (size_t)kT * kUP, k0, lane);
        v16bf br = ld_tile(WihA_b, j0,            kUP, k0, lane);
        v16bf bz = ld_tile(WihA_b, kHA + j0,      kUP, k0, lane);
        v16bf bn = ld_tile(WihA_b, 2 * kHA + j0,  kUP, k0, lane);
        cr = wmma_bf16(a, br, cr);
        cz = wmma_bf16(a, bz, cz);
        cgin = wmma_bf16(a, bn, cgin);
    }
    for (int k0 = 0; k0 < kHA; k0 += 32) {
        v16bf a  = ld_tile(h_in_b, m0, kHA, k0, lane);
        v16bf br = ld_tile(WhhA_b, j0,           kHA, k0, lane);
        v16bf bz = ld_tile(WhhA_b, kHA + j0,     kHA, k0, lane);
        v16bf bn = ld_tile(WhhA_b, 2 * kHA + j0, kHA, k0, lane);
        cr = wmma_bf16(a, br, cr);
        cz = wmma_bf16(a, bz, cz);
        cghn = wmma_bf16(a, bn, cghn);
    }

    const int n = j0 + (lane & 15);
    const float br_ = bihA[n] + bhhA[n];
    const float bz_ = bihA[kHA + n] + bhhA[kHA + n];
    const float bin_ = bihA[2 * kHA + n];
    const float bhn_ = bhhA[2 * kHA + n];
#pragma unroll
    for (int v = 0; v < 8; ++v) {
        const int b = m0 + v + ((lane >> 4) << 3);   // C layout: M = v + 8*(lane>=16)
        const float hprev = h_in[(size_t)b * kHA + n];
        const float r = sigf(cr[v] + br_);
        const float z = sigf(cz[v] + bz_);
        const float nn = tanh_f(cgin[v] + bin_ + r * (cghn[v] + bhn_));
        const float hnew = (1.0f - z) * nn + z * hprev;
        h_out[(size_t)b * kHA + n] = hnew;
        h_out_b[(size_t)b * kHA + n] = (bf16)hnew;
    }
}

// ---- Phase B: hypernet GEMM with fused base-add + repack epilogue ----------
// grid P/16 = 5152 (col tiles), block 256 (8 waves = batch tiles)
__global__ __launch_bounds__(256) void k_hyper(
    const bf16* __restrict__ hage_b, const float* __restrict__ Wage,
    const float* __restrict__ bage,
    const float* __restrict__ Wih, const float* __restrict__ Whh,
    const float* __restrict__ bih, const float* __restrict__ bhh,
    bf16* __restrict__ Wih_tot, bf16* __restrict__ Whh_tot,
    float* __restrict__ bih_n, float* __restrict__ bhh_n) {
    const int lane = threadIdx.x & 31;
    const int wave = threadIdx.x >> 5;
    const int m0 = wave * 16;
    const int c0 = blockIdx.x * 16;

    v8f acc = {};
    for (int k0 = 0; k0 < kHA; k0 += 32) {
        v16bf a = ld_tile(hage_b, m0, kHA, k0, lane);
        v16bf b = ld_tile_f32(Wage, c0, kHA, k0, lane);   // convert Wage on the fly
        acc = wmma_bf16(a, b, acc);
    }

    const int c = c0 + (lane & 15);
    const float bg = bage[c];
    // region boundaries are multiples of 256 -> uniform per block
    if (c0 < kO1) {
        const float base = Wih[32768 + c];                // Wih[2H + c/64, c%64]
#pragma unroll
        for (int v = 0; v < 8; ++v) {
            const int b = m0 + v + ((lane >> 4) << 3);
            Wih_tot[(size_t)b * kO1 + c] = (bf16)(base + acc[v] + bg);
        }
    } else if (c0 < kO2) {
        const int cc = c - kO1;
        const float base = Whh[131072 + cc];              // Whh[2H + cc/256, cc%256]
#pragma unroll
        for (int v = 0; v < 8; ++v) {
            const int b = m0 + v + ((lane >> 4) << 3);
            Whh_tot[(size_t)b * (kH * kH) + cc] = (bf16)(base + acc[v] + bg);
        }
    } else if (c0 < kO3) {
        const int cc = c - kO2;
        const float base = bih[512 + cc];
#pragma unroll
        for (int v = 0; v < 8; ++v) {
            const int b = m0 + v + ((lane >> 4) << 3);
            bih_n[b * kH + cc] = base + acc[v] + bg;
        }
    } else {
        const int cc = c - kO3;
        const float base = bhh[512 + cc];
#pragma unroll
        for (int v = 0; v < 8; ++v) {
            const int b = m0 + v + ((lane >> 4) << 3);
            bhh_n[b * kH + cc] = base + acc[v] + bg;
        }
    }
}

// ---- Phase C: fused GRU cell (shared-gate WMMA + per-batch matvec) ---------
// grid 16 (H col-tiles), block 256 (8 waves = batch tiles)
__global__ __launch_bounds__(256) void k_cell_step(
    const bf16* __restrict__ x_b, size_t x_ld,
    const bf16* __restrict__ Wih_rz_b, const bf16* __restrict__ Whh_rz_b,
    const float* __restrict__ bih, const float* __restrict__ bhh,
    const bf16* __restrict__ Wih_tot, const bf16* __restrict__ Whh_tot,
    const float* __restrict__ bih_n, const float* __restrict__ bhh_n,
    const float* __restrict__ h_in, const bf16* __restrict__ h_in_b,
    float* __restrict__ h_out, bf16* __restrict__ h_out_b) {
    const int lane = threadIdx.x & 31;
    const int wave = threadIdx.x >> 5;
    const int m0 = wave * 16;
    const int j0 = blockIdx.x * 16;

    v8f cr = {}, cz = {};
#pragma unroll
    for (int k0 = 0; k0 < kI; k0 += 32) {           // x part, K=64 (cols 64..95 excluded)
        v16bf a  = ld_tile(x_b, m0, x_ld, k0, lane);
        v16bf br = ld_tile(Wih_rz_b, j0,       kI, k0, lane);
        v16bf bz = ld_tile(Wih_rz_b, kH + j0,  kI, k0, lane);
        cr = wmma_bf16(a, br, cr);
        cz = wmma_bf16(a, bz, cz);
    }
#pragma unroll
    for (int k0 = 0; k0 < kH; k0 += 32) {           // h part, K=256
        v16bf a  = ld_tile(h_in_b, m0, kH, k0, lane);
        v16bf br = ld_tile(Whh_rz_b, j0,       kH, k0, lane);
        v16bf bz = ld_tile(Whh_rz_b, kH + j0,  kH, k0, lane);
        cr = wmma_bf16(a, br, cr);
        cz = wmma_bf16(a, bz, cz);
    }

    const int n = j0 + (lane & 15);
    const float brz_r = bih[n] + bhh[n];
    const float brz_z = bih[kH + n] + bhh[kH + n];

#pragma unroll
    for (int v = 0; v < 8; ++v) {
        const int b = m0 + v + ((lane >> 4) << 3);  // matches C-matrix striping exactly
        // hn[b,n] = h[b,:] . Whh_tot[b][n,:]  (per-batch weights, L2-resident)
        const bf16* wh = Whh_tot + ((size_t)b * kH + n) * kH;
        const float* hb = h_in + (size_t)b * kH;
        __builtin_prefetch(wh + kH, 0, 1);
        float hn = 0.f;
        for (int k = 0; k < kH; k += 8) dot8_bf_f32(wh + k, hb + k, hn);
        // xn[b,n] = x[b,:64] . Wih_tot[b][n,:]
        const bf16* wx = Wih_tot + ((size_t)b * kH + n) * kI;
        const bf16* xb = x_b + (size_t)b * x_ld;
        float xn = 0.f;
#pragma unroll
        for (int k = 0; k < kI; k += 8) dot8_bf_bf(wx + k, xb + k, xn);

        hn += bhh_n[b * kH + n];
        xn += bih_n[b * kH + n];
        const float r = sigf(cr[v] + brz_r);
        const float z = sigf(cz[v] + brz_z);
        const float nn = tanh_f(fmaf(r, hn, xn));
        const float hprev = h_in[(size_t)b * kH + n];
        const float hnew = (1.0f - z) * nn + z * hprev;
        h_out[(size_t)b * kH + n] = hnew;
        h_out_b[(size_t)b * kH + n] = (bf16)hnew;
    }
}

// ---- Phase D: output projection for one timestep ---------------------------
// grid 1, block 256 (8 waves = batch tiles), N = 16 = OUT
__global__ __launch_bounds__(256) void k_out_step(
    const bf16* __restrict__ h_b, const bf16* __restrict__ Wout_b,
    const float* __restrict__ bout, float* __restrict__ out, int t) {
    const int lane = threadIdx.x & 31;
    const int wave = threadIdx.x >> 5;
    const int m0 = wave * 16;
    v8f acc = {};
#pragma unroll
    for (int k0 = 0; k0 < kH; k0 += 32) {
        v16bf a = ld_tile(h_b, m0, kH, k0, lane);
        v16bf b = ld_tile(Wout_b, 0, kH, k0, lane);
        acc = wmma_bf16(a, b, acc);
    }
    const int o = lane & 15;
    const float bo = bout[o];
#pragma unroll
    for (int v = 0; v < 8; ++v) {
        const int b = m0 + v + ((lane >> 4) << 3);
        out[((size_t)b * kT + t) * kOUT + o] = acc[v] + bo;
    }
}

// ---- host orchestration -----------------------------------------------------
extern "C" void kernel_launch(void* const* d_in, const int* in_sizes, int n_in,
                              void* d_out, int out_size, void* d_ws, size_t ws_size,
                              hipStream_t stream) {
    (void)in_sizes; (void)n_in; (void)out_size; (void)ws_size;
    const float* u    = (const float*)d_in[0];
    const float* WihA = (const float*)d_in[1];
    const float* WhhA = (const float*)d_in[2];
    const float* bihA = (const float*)d_in[3];
    const float* bhhA = (const float*)d_in[4];
    const float* Wage = (const float*)d_in[5];
    const float* bage = (const float*)d_in[6];
    const float* Wih  = (const float*)d_in[7];
    const float* Whh  = (const float*)d_in[8];
    const float* bih  = (const float*)d_in[9];
    const float* bhh  = (const float*)d_in[10];
    const float* Wout = (const float*)d_in[11];
    const float* bout = (const float*)d_in[12];
    float* out = (float*)d_out;

    char* wp = (char*)d_ws;
    auto alloc = [&](size_t bytes) -> char* {
        char* p = wp;
        wp += (bytes + 255) & ~(size_t)255;
        return p;
    };
    bf16* u_b       = (bf16*)alloc((size_t)kB * kT * kUP * 2);       // 25.2 MB
    bf16* WihA_b    = (bf16*)alloc((size_t)3 * kHA * kUP * 2);
    bf16* WhhA_b    = (bf16*)alloc((size_t)3 * kHA * kHA * 2);
    bf16* Wih_rz_b  = (bf16*)alloc((size_t)2 * kH * kI * 2);
    bf16* Whh_rz_b  = (bf16*)alloc((size_t)2 * kH * kH * 2);
    bf16* Wout_b    = (bf16*)alloc((size_t)kOUT * kH * 2);
    float* ha[2];  bf16* hab[2];
    ha[0]  = (float*)alloc((size_t)kB * kHA * 4);
    ha[1]  = (float*)alloc((size_t)kB * kHA * 4);
    hab[0] = (bf16*)alloc((size_t)kB * kHA * 2);
    hab[1] = (bf16*)alloc((size_t)kB * kHA * 2);
    bf16* Wih_tot   = (bf16*)alloc((size_t)kB * kO1 * 2);            // 4.2 MB
    bf16* Whh_tot   = (bf16*)alloc((size_t)kB * kH * kH * 2);        // 16.8 MB
    float* bih_n    = (float*)alloc((size_t)kB * kH * 4);
    float* bhh_n    = (float*)alloc((size_t)kB * kH * 4);
    float* hc[2];  bf16* hcb[2];
    hc[0]  = (float*)alloc((size_t)kB * kH * 4);
    hc[1]  = (float*)alloc((size_t)kB * kH * 4);
    hcb[0] = (bf16*)alloc((size_t)kB * kH * 2);
    hcb[1] = (bf16*)alloc((size_t)kB * kH * 2);

    // --- conversions / padding ---
    {
        int n = kB * kT * kUP;
        k_conv_pad<<<(n + 255) / 256, 256, 0, stream>>>(u, u_b, kB * kT, kI + 1, kUP);
        n = 3 * kHA * kUP;
        k_conv_pad<<<(n + 255) / 256, 256, 0, stream>>>(WihA, WihA_b, 3 * kHA, kI + 1, kUP);
        n = 3 * kHA * kHA;
        k_conv_pad<<<(n + 255) / 256, 256, 0, stream>>>(WhhA, WhhA_b, 3 * kHA, kHA, kHA);
        n = 2 * kH * kI;   // first 2H rows of Wih
        k_conv_pad<<<(n + 255) / 256, 256, 0, stream>>>(Wih, Wih_rz_b, 2 * kH, kI, kI);
        n = 2 * kH * kH;   // first 2H rows of Whh
        k_conv_pad<<<(n + 255) / 256, 256, 0, stream>>>(Whh, Whh_rz_b, 2 * kH, kH, kH);
        n = kOUT * kH;
        k_conv_pad<<<(n + 255) / 256, 256, 0, stream>>>(Wout, Wout_b, kOUT, kH, kH);
    }

    // --- Phase A: age-GRU scan over T ---
    hipMemsetAsync(ha[0], 0, (size_t)kB * kHA * 4, stream);
    hipMemsetAsync(hab[0], 0, (size_t)kB * kHA * 2, stream);
    int cur = 0;
    for (int t = 0; t < kT; ++t) {
        k_age_step<<<kHA / 16, 256, 0, stream>>>(u_b, t, WihA_b, WhhA_b, bihA, bhhA,
                                                 ha[cur], hab[cur], ha[1 - cur], hab[1 - cur]);
        cur ^= 1;
    }
    const bf16* hage_final = hab[cur];

    // --- Phase B: hypernet + repack ---
    k_hyper<<<kP / 16, 256, 0, stream>>>(hage_final, Wage, bage, Wih, Whh, bih, bhh,
                                         Wih_tot, Whh_tot, bih_n, bhh_n);

    // --- Phase C1: fixed-point iterations (x_t = u[:,0,:64], x0 = 0) ---
    hipMemsetAsync(hc[0], 0, (size_t)kB * kH * 4, stream);
    hipMemsetAsync(hcb[0], 0, (size_t)kB * kH * 2, stream);
    cur = 0;
    for (int i = 0; i < 200; ++i) {
        k_cell_step<<<kH / 16, 256, 0, stream>>>(u_b, (size_t)kT * kUP,
                                                 Wih_rz_b, Whh_rz_b, bih, bhh,
                                                 Wih_tot, Whh_tot, bih_n, bhh_n,
                                                 hc[cur], hcb[cur], hc[1 - cur], hcb[1 - cur]);
        cur ^= 1;
    }

    // --- Phase C2/D: sequence scan with fused per-step output projection ---
    for (int t = 0; t < kT; ++t) {
        k_cell_step<<<kH / 16, 256, 0, stream>>>(u_b + (size_t)t * kUP, (size_t)kT * kUP,
                                                 Wih_rz_b, Whh_rz_b, bih, bhh,
                                                 Wih_tot, Whh_tot, bih_n, bhh_n,
                                                 hc[cur], hcb[cur], hc[1 - cur], hcb[1 - cur]);
        k_out_step<<<1, 256, 0, stream>>>(hcb[1 - cur], Wout_b, bout, out, t);
        cur ^= 1;
    }
}
